// ResClassFormer_39187281609175
// MI455X (gfx1250) — matrix operand order repaired and checked by reading, not compiled, not run on hardware
//
#include <hip/hip_runtime.h>
#include <hip/hip_fp16.h>
#include <math.h>

typedef __attribute__((ext_vector_type(16))) _Float16 v16h;
typedef __attribute__((ext_vector_type(8)))  _Float16 v8h;
typedef __attribute__((ext_vector_type(8)))  float    v8f;

#define BATCH 64
#define NTOK  784
#define DIM   256
#define HEADS 8
#define HD    32
#define IMG   28
#define NKV   265      // 196 + 49 + 16 + 4
#define NKVP  288      // padded to 9*32 for WMMA K loop

// ---------------- tiled WMMA GEMM: C = A(MxK,f16) * Bt(NxK,f16)^T + bias ----
#define BM 128
#define BN 128
#define BK 32
#define LDA 40   // padded LDS stride (halves) -> conflict-free b128 frag loads
#define LDB 40

union U16h { v16h v; v8h h2[2]; };

// ------- gfx1250 async global->LDS path (guarded; falls back to ld+ds_store) --
#if __has_builtin(__builtin_amdgcn_global_load_async_to_lds_b128)
#define HAVE_ASYNC_LDS 1
typedef int v4i __attribute__((vector_size(16)));
typedef __attribute__((address_space(1))) v4i* gv4p;   // global src
typedef __attribute__((address_space(3))) v4i* lv4p;   // LDS dst
__device__ inline void async_cp16(const _Float16* g, _Float16* l) {
    // AMDGCN: generic addr of global mem == AS1 addr; low 32 bits of a generic
    // LDS addr == raw LDS offset, so inttoptr conversions are exact.
    __builtin_amdgcn_global_load_async_to_lds_b128(
        (gv4p)(size_t)g, (lv4p)(unsigned)(size_t)l, 0, 0);
}
__device__ inline void async_wait() {
#if __has_builtin(__builtin_amdgcn_s_wait_asynccnt)
    __builtin_amdgcn_s_wait_asynccnt(0);
#else
    asm volatile("s_wait_asynccnt 0" ::: "memory");
#endif
}
#else
#define HAVE_ASYNC_LDS 0
#endif

__device__ inline void tile_cp16(const _Float16* g, _Float16* l) {
#if HAVE_ASYNC_LDS
    async_cp16(g, l);
#else
    *(v8h*)l = *(const v8h*)g;
#endif
}

__device__ inline float gelu_exact(float x) {
    return 0.5f * x * (1.0f + erff(x * 0.70710678118654752f));
}

// EPI: 0 = store f16 (+bias), 1 = gelu -> f16, 2 = +bias +resid -> f32
template<int EPI>
__global__ __launch_bounds__(256)
void gemm_f16(const _Float16* __restrict__ A, const _Float16* __restrict__ Bt,
              const float* __restrict__ bias, const float* __restrict__ resid,
              _Float16* __restrict__ outH, float* __restrict__ outF,
              int M, int N, int K) {
    __shared__ _Float16 As[BM * LDA];
    __shared__ _Float16 Bs[BN * LDB];
    const int tid  = threadIdx.x;
    const int lane = tid & 31;
    const int wave = tid >> 5;                 // 0..7
    const int wm = wave >> 2, wn = wave & 3;   // 2x4 wave grid -> 64x32 per wave
    const int m_blk = blockIdx.y * BM;
    const int n_blk = blockIdx.x * BN;
    const int kb  = (lane >> 4) * 8;
    const int kb2 = (lane >> 4) * 16;
    const int lr = tid >> 1, lc = (tid & 1) * 16;   // tile-loader coords

    v8f acc[4][2] = {};
    for (int k0 = 0; k0 < K; k0 += BK) {
        { // A tile: 128x32 halves, 16 halves/thread
            int gr = m_blk + lr;
            if (gr < M) {
                const _Float16* p = A + (size_t)gr * K + k0 + lc;
                tile_cp16(p,     &As[lr * LDA + lc]);
                tile_cp16(p + 8, &As[lr * LDA + lc + 8]);
                if (k0 + BK < K) __builtin_prefetch(p + BK, 0, 0);
            } else {
                *(v8h*)&As[lr * LDA + lc]     = (v8h){};
                *(v8h*)&As[lr * LDA + lc + 8] = (v8h){};
            }
        }
        { // B tile (Bt is N x K row-major): 128x32 halves, 16 halves/thread
            const _Float16* p = Bt + (size_t)(n_blk + lr) * K + k0 + lc;
            tile_cp16(p,     &Bs[lr * LDB + lc]);
            tile_cp16(p + 8, &Bs[lr * LDB + lc + 8]);
        }
#if HAVE_ASYNC_LDS
        async_wait();
#endif
        __syncthreads();
        U16h af[4], bf[2];
        #pragma unroll
        for (int i = 0; i < 4; i++) {
            int r = wm * 64 + i * 16 + (lane & 15);
            af[i].h2[0] = *(const v8h*)&As[r * LDA + kb2];      // K = kb2..kb2+7
            af[i].h2[1] = *(const v8h*)&As[r * LDA + kb2 + 8];  // K = kb2+8..kb2+15
        }
        #pragma unroll
        for (int j = 0; j < 2; j++) {
            int r = wn * 32 + j * 16 + (lane & 15);
            bf[j].h2[0] = *(const v8h*)&Bs[r * LDB + kb2];
            bf[j].h2[1] = *(const v8h*)&Bs[r * LDB + kb2 + 8];
        }
        #pragma unroll
        for (int i = 0; i < 4; i++)
            #pragma unroll
            for (int j = 0; j < 2; j++)
                acc[i][j] = __builtin_amdgcn_wmma_f32_16x16x32_f16(
                    false, af[i].v, false, bf[j].v, (short)0, acc[i][j], false, false);
        __syncthreads();
    }
    #pragma unroll
    for (int i = 0; i < 4; i++)
        #pragma unroll
        for (int j = 0; j < 2; j++)
            #pragma unroll
            for (int g = 0; g < 8; g++) {
                int gm = m_blk + wm * 64 + i * 16 + (lane >> 4) * 8 + g;
                int gn = n_blk + wn * 32 + j * 16 + (lane & 15);
                if (gm >= M || gn >= N) continue;
                float v = acc[i][j][g] + bias[gn];
                if (EPI == 0) {
                    outH[(size_t)gm * N + gn] = (_Float16)v;
                } else if (EPI == 1) {
                    outH[(size_t)gm * N + gn] = (_Float16)gelu_exact(v);
                } else {
                    outF[(size_t)gm * N + gn] = v + resid[(size_t)gm * N + gn];
                }
            }
}

// --------- implicit-GEMM 3x3 conv, channels-last: xp(b,p,128) += q(b,p',256)*W ---
__global__ __launch_bounds__(256)
void conv_gemm(const _Float16* __restrict__ q, const _Float16* __restrict__ wT,
               float* __restrict__ xp) {
    __shared__ _Float16 As[BM * LDA];
    __shared__ _Float16 Bs[BN * LDB];
    const int tid  = threadIdx.x;
    const int lane = tid & 31;
    const int wave = tid >> 5;
    const int wm = wave >> 2, wn = wave & 3;
    const int m_blk = blockIdx.y * BM;
    const int kb  = (lane >> 4) * 8;
    const int kb2 = (lane >> 4) * 16;
    const int lr = tid >> 1, lc = (tid & 1) * 16;

    v8f acc[4][2] = {};
    for (int tap = 0; tap < 9; tap++) {
        const int dy = tap / 3 - 1, dx = tap % 3 - 1;
        for (int k0 = 0; k0 < DIM; k0 += BK) {
            { // im2col A tile (zero padding at image borders)
                int gm = m_blk + lr;
                int b = gm / NTOK, p = gm % NTOK;
                int y = p / IMG + dy, x = p % IMG + dx;
                if (y >= 0 && y < IMG && x >= 0 && x < IMG) {
                    const _Float16* pp = q + ((size_t)(b * NTOK) + y * IMG + x) * DIM + k0 + lc;
                    tile_cp16(pp,     &As[lr * LDA + lc]);
                    tile_cp16(pp + 8, &As[lr * LDA + lc + 8]);
                } else {
                    *(v8h*)&As[lr * LDA + lc]     = (v8h){};
                    *(v8h*)&As[lr * LDA + lc + 8] = (v8h){};
                }
            }
            { // B tile from per-tap transposed weight (tap,oc,ic)
                const _Float16* p = wT + ((size_t)(tap * 128 + lr)) * DIM + k0 + lc;
                tile_cp16(p,     &Bs[lr * LDB + lc]);
                tile_cp16(p + 8, &Bs[lr * LDB + lc + 8]);
            }
#if HAVE_ASYNC_LDS
            async_wait();
#endif
            __syncthreads();
            U16h af[4], bf[2];
            #pragma unroll
            for (int i = 0; i < 4; i++) {
                int r = wm * 64 + i * 16 + (lane & 15);
                af[i].h2[0] = *(const v8h*)&As[r * LDA + kb2];
                af[i].h2[1] = *(const v8h*)&As[r * LDA + kb2 + 8];
            }
            #pragma unroll
            for (int j = 0; j < 2; j++) {
                int r = wn * 32 + j * 16 + (lane & 15);
                bf[j].h2[0] = *(const v8h*)&Bs[r * LDB + kb2];
                bf[j].h2[1] = *(const v8h*)&Bs[r * LDB + kb2 + 8];
            }
            #pragma unroll
            for (int i = 0; i < 4; i++)
                #pragma unroll
                for (int j = 0; j < 2; j++)
                    acc[i][j] = __builtin_amdgcn_wmma_f32_16x16x32_f16(
                        false, af[i].v, false, bf[j].v, (short)0, acc[i][j], false, false);
            __syncthreads();
        }
    }
    #pragma unroll
    for (int i = 0; i < 4; i++)
        #pragma unroll
        for (int j = 0; j < 2; j++)
            #pragma unroll
            for (int g = 0; g < 8; g++) {
                int gm = m_blk + wm * 64 + i * 16 + (lane >> 4) * 8 + g;
                int gn = wn * 32 + j * 16 + (lane & 15);
                if (gn < 128) xp[(size_t)gm * 128 + gn] = acc[i][j][g];
            }
}

// ---------------- layernorm over 256 (one wave per row) -> f16 ----------------
__global__ __launch_bounds__(256)
void ln256(const float* __restrict__ X, const float* __restrict__ g,
           const float* __restrict__ b, _Float16* __restrict__ out,
           int M, float eps) {
    int lane = threadIdx.x & 31;
    int row  = blockIdx.x * 8 + (threadIdx.x >> 5);
    if (row >= M) return;
    const float* xr = X + (size_t)row * DIM;
    float v[8], s = 0.f;
    #pragma unroll
    for (int i = 0; i < 8; i++) { v[i] = xr[lane + i * 32]; s += v[i]; }
    #pragma unroll
    for (int o = 16; o; o >>= 1) s += __shfl_xor(s, o, 32);
    float mean = s * (1.0f / 256.0f), vs = 0.f;
    #pragma unroll
    for (int i = 0; i < 8; i++) { float d = v[i] - mean; vs += d * d; }
    #pragma unroll
    for (int o = 16; o; o >>= 1) vs += __shfl_xor(vs, o, 32);
    float inv = rsqrtf(vs * (1.0f / 256.0f) + eps);
    #pragma unroll
    for (int i = 0; i < 8; i++) {
        int c = lane + i * 32;
        out[(size_t)row * DIM + c] = (_Float16)((v[i] - mean) * inv * g[c] + b[c]);
    }
}

// ------------- LN(128) + GELU + channel mean/max -> amap(b,2,784) -------------
__global__ __launch_bounds__(256)
void lnp_amap(const float* __restrict__ xp, const float* __restrict__ g,
              const float* __restrict__ b, float* __restrict__ amap) {
    int lane = threadIdx.x & 31;
    int pix  = blockIdx.x * 8 + (threadIdx.x >> 5);   // b*784 + p
    const float* xr = xp + (size_t)pix * 128;
    float v[4], s = 0.f;
    #pragma unroll
    for (int i = 0; i < 4; i++) { v[i] = xr[lane + i * 32]; s += v[i]; }
    #pragma unroll
    for (int o = 16; o; o >>= 1) s += __shfl_xor(s, o, 32);
    float mean = s * (1.0f / 128.0f), vs = 0.f;
    #pragma unroll
    for (int i = 0; i < 4; i++) { float d = v[i] - mean; vs += d * d; }
    #pragma unroll
    for (int o = 16; o; o >>= 1) vs += __shfl_xor(vs, o, 32);
    float inv = rsqrtf(vs * (1.0f / 128.0f) + 1e-5f);
    float sm = 0.f, mx = -3.0e38f;
    #pragma unroll
    for (int i = 0; i < 4; i++) {
        int c = lane + i * 32;
        float ge = gelu_exact((v[i] - mean) * inv * g[c] + b[c]);
        sm += ge; mx = fmaxf(mx, ge);
    }
    #pragma unroll
    for (int o = 16; o; o >>= 1) { sm += __shfl_xor(sm, o, 32); mx = fmaxf(mx, __shfl_xor(mx, o, 32)); }
    if (lane == 0) {
        int bb = pix / NTOK, p = pix % NTOK;
        amap[((size_t)bb * 2 + 0) * NTOK + p] = sm * (1.0f / 128.0f);
        amap[((size_t)bb * 2 + 1) * NTOK + p] = mx;
    }
}

// -------- per-window: tiny 2->1ch 3x3 conv + sigmoid, argmax -> flat idx -------
__global__ void window_argmax(const float* __restrict__ amap,
                              const float* __restrict__ att_ws,
                              int* __restrict__ idxb) {
    int t = blockIdx.x * blockDim.x + threadIdx.x;
    if (t >= BATCH * NKV) return;
    int b = t / NKV, s = t % NKV;
    int ri, r, wy, wx;
    if (s < 196)      { ri = 0; r = 2;  int w = s;       wy = w / 14; wx = w % 14; }
    else if (s < 245) { ri = 1; r = 4;  int w = s - 196; wy = w / 7;  wx = w % 7;  }
    else if (s < 261) { ri = 2; r = 7;  int w = s - 245; wy = w / 4;  wx = w % 4;  }
    else              { ri = 3; r = 14; int w = s - 261; wy = w / 2;  wx = w % 2;  }
    const float* wts = att_ws + ri * 18;            // (1,2,3,3)
    const float* am  = amap + (size_t)b * 2 * NTOK;
    float best = -3.0e38f; int bi = 0;
    for (int dyy = 0; dyy < r; dyy++)
        for (int dxx = 0; dxx < r; dxx++) {
            int y = wy * r + dyy, x = wx * r + dxx;
            float a = 0.f;
            for (int ic = 0; ic < 2; ic++)
                for (int ky = 0; ky < 3; ky++)
                    for (int kx = 0; kx < 3; kx++) {
                        int yy = y + ky - 1, xx = x + kx - 1;
                        if (yy >= 0 && yy < IMG && xx >= 0 && xx < IMG)
                            a += am[ic * NTOK + yy * IMG + xx] * wts[(ic * 3 + ky) * 3 + kx];
                    }
            a = 1.0f / (1.0f + expf(-a));
            if (a > best) { best = a; bi = y * IMG + x; }
        }
    idxb[t] = bi;
}

// ------------------------- gather 265 kv tokens from q ------------------------
__global__ void gather_kv(const _Float16* __restrict__ q, const int* __restrict__ idxb,
                          _Float16* __restrict__ kvtok) {
    int t = blockIdx.x * blockDim.x + threadIdx.x;    // (b*265+s)*32 + c8
    if (t >= BATCH * NKV * 32) return;
    int c8 = t & 31, bs = t >> 5;
    int b = bs / NKV;
    int p = idxb[bs];
    *(v8h*)(kvtok + (size_t)bs * DIM + c8 * 8) =
        *(const v8h*)(q + ((size_t)(b * NTOK + p)) * DIM + c8 * 8);
}

// -------- repack V into (b,h,hd,288) with zero-padded tokens [265,288) --------
__global__ void repack_v(const _Float16* __restrict__ kvout, _Float16* __restrict__ vT) {
    int t = blockIdx.x * blockDim.x + threadIdx.x;
    if (t >= BATCH * HEADS * HD * NKVP) return;
    int s = t % NKVP; int rest = t / NKVP;
    int hd = rest % HD; rest /= HD;
    int h = rest % HEADS; int b = rest / HEADS;
    _Float16 val = (_Float16)0.0f;
    if (s < NKV) val = kvout[((size_t)(b * NKV + s)) * 512 + 256 + h * HD + hd];
    vT[t] = val;
}

// -------- attention: one wave per (b,h,16 q-rows); S,P in LDS; WMMA both GEMMs
__global__ __launch_bounds__(32)
void attention(const _Float16* __restrict__ q, const _Float16* __restrict__ kvout,
               const _Float16* __restrict__ vT, _Float16* __restrict__ attnout) {
    __shared__ float    Ssh[16][280];
    __shared__ _Float16 Psh[16][NKVP];
    const int lane = threadIdx.x;
    const int mt = blockIdx.x, h = blockIdx.y, b = blockIdx.z;
    const int m0 = mt * 16;
    const int kb = (lane >> 4) * 8, kb2 = (lane >> 4) * 16;

    U16h aq;
    {
        const _Float16* p = q + ((size_t)(b * NTOK + m0 + (lane & 15))) * DIM + h * HD;
        aq.h2[0] = *(const v8h*)(p + kb2);
        aq.h2[1] = *(const v8h*)(p + kb2 + 8);
        const _Float16 sc = (_Float16)0.17677669529663687f;   // 32^-0.5
        #pragma unroll
        for (int i = 0; i < 16; i++) aq.v[i] = aq.v[i] * sc;
    }
    for (int nt = 0; nt < 17; nt++) {
        int s = nt * 16 + (lane & 15);
        U16h bk; bk.h2[0] = (v8h){}; bk.h2[1] = (v8h){};
        if (s < NKV) {
            const _Float16* p = kvout + ((size_t)(b * NKV + s)) * 512 + h * HD + kb2;
            bk.h2[0] = *(const v8h*)p;
            bk.h2[1] = *(const v8h*)(p + 8);
        }
        v8f c = {};
        c = __builtin_amdgcn_wmma_f32_16x16x32_f16(false, aq.v, false, bk.v, (short)0, c, false, false);
        #pragma unroll
        for (int g = 0; g < 8; g++)
            Ssh[g + (lane >> 4) * 8][nt * 16 + (lane & 15)] = c[g];
    }
    __syncthreads();
    { // softmax over 265: two lanes per row, strided
        int r = lane & 15, half = lane >> 4;
        float mx = -3.0e38f;
        for (int c = half; c < NKV; c += 2) mx = fmaxf(mx, Ssh[r][c]);
        mx = fmaxf(mx, __shfl_xor(mx, 16, 32));
        float sm = 0.f;
        for (int c = half; c < NKV; c += 2) { float e = expf(Ssh[r][c] - mx); Ssh[r][c] = e; sm += e; }
        sm += __shfl_xor(sm, 16, 32);
        float inv = 1.0f / sm;
        for (int c = half; c < NKV; c += 2)        Psh[r][c] = (_Float16)(Ssh[r][c] * inv);
        for (int c = NKV + half; c < NKVP; c += 2) Psh[r][c] = (_Float16)0.0f;
    }
    __syncthreads();
    v8f ao[2] = {};
    for (int kt = 0; kt < 9; kt++) {
        U16h ap;
        ap.h2[0] = *(const v8h*)&Psh[lane & 15][kt * 32 + kb2];
        ap.h2[1] = *(const v8h*)&Psh[lane & 15][kt * 32 + kb2 + 8];
        #pragma unroll
        for (int j = 0; j < 2; j++) {
            U16h bv;
            int hd = j * 16 + (lane & 15);
            const _Float16* p = vT + (((size_t)(b * HEADS + h) * HD + hd)) * NKVP + kt * 32 + kb2;
            bv.h2[0] = *(const v8h*)p;
            bv.h2[1] = *(const v8h*)(p + 8);
            ao[j] = __builtin_amdgcn_wmma_f32_16x16x32_f16(false, ap.v, false, bv.v, (short)0, ao[j], false, false);
        }
    }
    #pragma unroll
    for (int j = 0; j < 2; j++)
        #pragma unroll
        for (int g = 0; g < 8; g++) {
            int gm = m0 + g + (lane >> 4) * 8;
            int hd = j * 16 + (lane & 15);
            attnout[((size_t)(b * NTOK + gm)) * DIM + h * HD + hd] = (_Float16)ao[j][g];
        }
}

// --------------------------- weight preparation ------------------------------
__global__ void transpose_f16(const float* __restrict__ W, _Float16* __restrict__ Wt,
                              int K, int N) {   // W: KxN -> Wt: NxK
    int t = blockIdx.x * blockDim.x + threadIdx.x;
    if (t >= K * N) return;
    int n = t % N, k = t / N;
    Wt[(size_t)n * K + k] = (_Float16)W[t];
}

__global__ void conv_w_prep(const float* __restrict__ W, _Float16* __restrict__ Wt) {
    int t = blockIdx.x * blockDim.x + threadIdx.x;   // OIHW (128,256,3,3)
    if (t >= 128 * 256 * 9) return;
    int kx = t % 3; int rest = t / 3;
    int ky = rest % 3; rest /= 3;
    int ic = rest % 256; int oc = rest / 256;
    Wt[((size_t)((ky * 3 + kx) * 128 + oc)) * 256 + ic] = (_Float16)W[t];
}

// ------------------------------------------------------------------------------
extern "C" void kernel_launch(void* const* d_in, const int* in_sizes, int n_in,
                              void* d_out, int out_size, void* d_ws, size_t ws_size,
                              hipStream_t stream) {
    const float* x      = (const float*)d_in[0];
    const float* ln1_g  = (const float*)d_in[1];
    const float* ln1_b  = (const float*)d_in[2];
    const float* ln2_g  = (const float*)d_in[3];
    const float* ln2_b  = (const float*)d_in[4];
    const float* wq     = (const float*)d_in[5];
    const float* bq     = (const float*)d_in[6];
    const float* wkv    = (const float*)d_in[7];
    const float* bkv    = (const float*)d_in[8];
    const float* wo     = (const float*)d_in[9];
    const float* bo     = (const float*)d_in[10];
    const float* w1     = (const float*)d_in[11];
    const float* b1     = (const float*)d_in[12];
    const float* w2     = (const float*)d_in[13];
    const float* b2     = (const float*)d_in[14];
    const float* convw  = (const float*)d_in[15];
    const float* lnp_g  = (const float*)d_in[16];
    const float* lnp_b  = (const float*)d_in[17];
    const float* att_ws = (const float*)d_in[18];

    const int M = BATCH * NTOK;          // 50176
    const int Mkv = BATCH * NKV;         // 16960
    char* ws = (char*)d_ws;
    size_t off = 0;
    auto alloc = [&](size_t bytes) -> char* {
        char* p = ws + off;
        off += (bytes + 255) & ~(size_t)255;
        return p;
    };
    _Float16* wqT   = (_Float16*)alloc((size_t)256 * 256 * 2);
    _Float16* wkvT  = (_Float16*)alloc((size_t)512 * 256 * 2);
    _Float16* woT   = (_Float16*)alloc((size_t)256 * 256 * 2);
    _Float16* w1T   = (_Float16*)alloc((size_t)1024 * 256 * 2);
    _Float16* w2T   = (_Float16*)alloc((size_t)256 * 1024 * 2);
    _Float16* wcT   = (_Float16*)alloc((size_t)9 * 128 * 256 * 2);
    _Float16* xn1   = (_Float16*)alloc((size_t)M * DIM * 2);
    _Float16* qf16  = (_Float16*)alloc((size_t)M * DIM * 2);
    float*    xp    = (float*)   alloc((size_t)M * 128 * 4);
    float*    amap  = (float*)   alloc((size_t)BATCH * 2 * NTOK * 4);
    int*      idxb  = (int*)     alloc((size_t)BATCH * NKV * 4);
    _Float16* kvtok = (_Float16*)alloc((size_t)Mkv * DIM * 2);
    _Float16* kvout = (_Float16*)alloc((size_t)Mkv * 512 * 2);
    _Float16* vT    = (_Float16*)alloc((size_t)BATCH * HEADS * HD * NKVP * 2);
    _Float16* aout  = (_Float16*)alloc((size_t)M * DIM * 2);
    float*    x2    = (float*)   alloc((size_t)M * DIM * 4);
    _Float16* xn2   = (_Float16*)alloc((size_t)M * DIM * 2);
    _Float16* ff1   = (_Float16*)alloc((size_t)M * 1024 * 2);
    (void)ws_size; (void)in_sizes; (void)n_in; (void)out_size;

    // weight prep
    transpose_f16<<<(256 * 256 + 255) / 256, 256, 0, stream>>>(wq, wqT, 256, 256);
    transpose_f16<<<(256 * 512 + 255) / 256, 256, 0, stream>>>(wkv, wkvT, 256, 512);
    transpose_f16<<<(256 * 256 + 255) / 256, 256, 0, stream>>>(wo, woT, 256, 256);
    transpose_f16<<<(256 * 1024 + 255) / 256, 256, 0, stream>>>(w1, w1T, 256, 1024);
    transpose_f16<<<(1024 * 256 + 255) / 256, 256, 0, stream>>>(w2, w2T, 1024, 256);
    conv_w_prep<<<(128 * 256 * 9 + 255) / 256, 256, 0, stream>>>(convw, wcT);

    // LN1 + Q projection
    ln256<<<M / 8, 256, 0, stream>>>(x, ln1_g, ln1_b, xn1, M, 1e-6f);
    gemm_f16<0><<<dim3(256 / BN, M / BM), 256, 0, stream>>>(
        xn1, wqT, bq, nullptr, qf16, nullptr, M, 256, 256);

    // find_kv: conv -> LN+GELU+amap -> window argmax -> gather
    conv_gemm<<<dim3(1, M / BM), 256, 0, stream>>>(qf16, wcT, xp);
    lnp_amap<<<M / 8, 256, 0, stream>>>(xp, lnp_g, lnp_b, amap);
    window_argmax<<<(BATCH * NKV + 255) / 256, 256, 0, stream>>>(amap, att_ws, idxb);
    gather_kv<<<(BATCH * NKV * 32 + 255) / 256, 256, 0, stream>>>(qf16, idxb, kvtok);

    // KV projection, V repack, attention
    gemm_f16<0><<<dim3(512 / BN, (Mkv + BM - 1) / BM), 256, 0, stream>>>(
        kvtok, wkvT, bkv, nullptr, kvout, nullptr, Mkv, 512, 256);
    repack_v<<<(BATCH * HEADS * HD * NKVP + 255) / 256, 256, 0, stream>>>(kvout, vT);
    attention<<<dim3(NTOK / 16, HEADS, BATCH), 32, 0, stream>>>(qf16, kvout, vT, aout);

    // out projection + residual
    gemm_f16<2><<<dim3(256 / BN, M / BM), 256, 0, stream>>>(
        aout, woT, bo, x, nullptr, x2, M, 256, 256);

    // LN2 + FFN
    ln256<<<M / 8, 256, 0, stream>>>(x2, ln2_g, ln2_b, xn2, M, 1e-6f);
    gemm_f16<1><<<dim3(1024 / BN, M / BM), 256, 0, stream>>>(
        xn2, w1T, b1, nullptr, ff1, nullptr, M, 1024, 256);
    gemm_f16<2><<<dim3(256 / BN, M / BM), 256, 0, stream>>>(
        ff1, w2T, b2, x2, nullptr, (float*)d_out, M, 256, 1024);
}
